// MLABlock_6545530159426
// MI455X (gfx1250) — 16-bit and 8-bit matrix operand order repaired and checked by reading, compiled, not run on hardware
//
#include <hip/hip_runtime.h>

typedef __bf16 bf16_t;
typedef bf16_t v16bf __attribute__((ext_vector_type(16)));
typedef float  v8f   __attribute__((ext_vector_type(8)));
typedef int    v16i  __attribute__((ext_vector_type(16)));
typedef int    v8i   __attribute__((ext_vector_type(8)));
typedef int    v4i_t __attribute__((vector_size(16)));

union FragB16 { unsigned int u[8];  v16bf v; };
union Frag16  { unsigned int u[16]; v16i  v; };
union Frag8   { unsigned int u[8];  v8i   v; };
union AccF    { float        f[8];  v8f   v; };

#define BATCH 2
#define SEQ   2048
#define HID   2048
#define NH    16
#define HD    128
#define MTOT  (BATCH * SEQ)
#define NQKV  (3 * HID)
#define FP8_MAX 448.0f

#define LKSTR 144
#define LASTR 80

#define AS1 __attribute__((address_space(1)))
#define AS3 __attribute__((address_space(3)))

#if __has_builtin(__builtin_amdgcn_global_load_async_to_lds_b128)
#define HAVE_ASYNC_LDS 1
__device__ __forceinline__ void async_cp16(void* lds, const void* g) {
    __builtin_amdgcn_global_load_async_to_lds_b128(
        (AS1 v4i_t*)g, (AS3 v4i_t*)lds, 0, 0);
}
#else
#define HAVE_ASYNC_LDS 0
#endif

__device__ __forceinline__ void wait_async() {
#if HAVE_ASYNC_LDS
#if __has_builtin(__builtin_amdgcn_s_wait_asynccnt)
    __builtin_amdgcn_s_wait_asynccnt(0);
#else
    asm volatile("s_wait_asynccnt 0x0" ::: "memory");
#endif
#endif
}

__device__ __forceinline__ unsigned short f32_to_bf16(float f) {
    unsigned int u = __float_as_uint(f);
    u += 0x7FFFu + ((u >> 16) & 1u);
    return (unsigned short)(u >> 16);
}
__device__ __forceinline__ float bf16_to_f32(unsigned short h) {
    return __uint_as_float(((unsigned int)h) << 16);
}
__device__ __forceinline__ unsigned int f32_to_e4m3(float x) {
    unsigned int u   = __float_as_uint(x);
    unsigned int sgn = (u >> 24) & 0x80u;
    float a = fabsf(x);
    if (!(a >= 1.953125e-3f)) return sgn;
    if (a >= FP8_MAX)         return sgn | 0x7Eu;
    unsigned int au = __float_as_uint(a);
    au += 0x0007FFFFu + ((au >> 20) & 1u);
    unsigned int e = ((au >> 23) & 0xFFu) - 127u + 7u;
    unsigned int m = (au >> 20) & 0x7u;
    return sgn | (e << 3) | m;
}
__device__ __forceinline__ unsigned int pk_e4m3(float a, float b) {
#if __has_builtin(__builtin_amdgcn_cvt_pk_fp8_f32)
    return ((unsigned int)__builtin_amdgcn_cvt_pk_fp8_f32(a, b, 0, false)) & 0xFFFFu;
#else
    return f32_to_e4m3(a) | (f32_to_e4m3(b) << 8);
#endif
}

__global__ void init_kernel(unsigned int* amax, float* scales) {
    if (threadIdx.x < 4) { amax[threadIdx.x] = 0u; scales[threadIdx.x] = 0.f; }
}

__global__ __launch_bounds__(256) void gemm_qkv_kernel(
    const float* __restrict__ X,
    const float* __restrict__ W,
    unsigned short* __restrict__ qb,
    unsigned short* __restrict__ kb,
    unsigned short* __restrict__ vb,
    unsigned int* __restrict__ amax)
{
    __shared__ unsigned int lA[128 * 17];
    __shared__ unsigned int lB[128 * 17];

    const int m0 = blockIdx.x * 128;
    const int n0 = blockIdx.y * 128;
    const int tid  = threadIdx.x;
    const int wave = tid >> 5;
    const int lane = tid & 31;
    const int hl   = lane >> 4;
    const int ln   = lane & 15;

    AccF acc[8];
#pragma unroll
    for (int nt = 0; nt < 8; ++nt)
#pragma unroll
        for (int j = 0; j < 8; ++j) acc[nt].f[j] = 0.f;

    for (int k0 = 0; k0 < HID; k0 += 32) {
#pragma unroll
        for (int j = 0; j < 8; ++j) {
            int p   = tid + 256 * j;
            int row = p >> 4;
            int kp  = p & 15;
            const float2 xv = *(const float2*)(X + (size_t)(m0 + row) * HID + k0 + kp * 2);
            unsigned int pk = (unsigned int)f32_to_bf16(xv.x) |
                              ((unsigned int)f32_to_bf16(xv.y) << 16);
            lA[row * 17 + kp] = pk;
        }
#pragma unroll
        for (int j = 0; j < 8; ++j) {
            int p  = tid + 256 * j;
            int n  = p & 127;
            int kp = p >> 7;
            float b0 = W[(size_t)(k0 + 2 * kp)     * NQKV + n0 + n];
            float b1 = W[(size_t)(k0 + 2 * kp + 1) * NQKV + n0 + n];
            lB[n * 17 + kp] = (unsigned int)f32_to_bf16(b0) |
                              ((unsigned int)f32_to_bf16(b1) << 16);
        }
        if (k0 + 32 < HID) {
            __builtin_prefetch(X + (size_t)(m0 + (tid >> 1)) * HID + k0 + 32 + (tid & 1) * 16, 0, 0);
            __builtin_prefetch(W + (size_t)(k0 + 32 + (tid >> 4)) * NQKV + n0 + (tid & 15) * 8, 0, 0);
        }
        __syncthreads();

        FragB16 af;
#pragma unroll
        for (int v = 0; v < 8; ++v) {
            int kp = (v & 3) + ((v >> 2) * 8) + hl * 4;
            af.u[v] = lA[(wave * 16 + ln) * 17 + kp];
        }
#pragma unroll
        for (int nt = 0; nt < 8; ++nt) {
            FragB16 bfg;
#pragma unroll
            for (int v = 0; v < 8; ++v) {
                int kp = (v & 3) + ((v >> 2) * 8) + hl * 4;
                bfg.u[v] = lB[(nt * 16 + ln) * 17 + kp];
            }
            acc[nt].v = __builtin_amdgcn_wmma_f32_16x16x32_bf16(
                false, af.v, false, bfg.v, (short)0, acc[nt].v, false, false);
        }
        __syncthreads();
    }

    const int which = n0 >> 11;
    const int h     = (n0 & 2047) >> 7;
    unsigned short* dst = (which == 0) ? qb : ((which == 1) ? kb : vb);
    float lamax = 0.f;
#pragma unroll
    for (int nt = 0; nt < 8; ++nt) {
#pragma unroll
        for (int v = 0; v < 8; ++v) {
            float c = acc[nt].f[v];
            lamax = fmaxf(lamax, fabsf(c));
            int m = m0 + wave * 16 + v + hl * 8;
            int d = nt * 16 + ln;
            int b = m >> 11, s = m & 2047;
            dst[(((size_t)(b * NH + h)) * SEQ + s) * HD + d] = f32_to_bf16(c);
        }
    }
#pragma unroll
    for (int msk = 16; msk >= 1; msk >>= 1)
        lamax = fmaxf(lamax, __shfl_xor(lamax, msk, 32));
    if (lane == 0) atomicMax(&amax[which], __float_as_uint(lamax));
}

__global__ __launch_bounds__(256) void quant_kernel(
    const unsigned short* __restrict__ qkv_bf,
    unsigned char* __restrict__ qkv8,
    const unsigned int* __restrict__ amax,
    float* __restrict__ scales)
{
    const size_t N1 = (size_t)BATCH * NH * SEQ * HD;
    size_t idx = ((size_t)blockIdx.x * blockDim.x + threadIdx.x) * 8;
    if (blockIdx.x == 0 && threadIdx.x < 3)
        scales[threadIdx.x] = __uint_as_float(amax[threadIdx.x]) * (1.0f / FP8_MAX);
    int which = (int)(idx / N1);
    float sc  = __uint_as_float(amax[which]) * (1.0f / FP8_MAX);
    float inv = (sc > 0.f) ? (1.0f / sc) : 0.f;
    float x[8];
#pragma unroll
    for (int j = 0; j < 8; ++j) x[j] = bf16_to_f32(qkv_bf[idx + j]) * inv;
    unsigned int lo = pk_e4m3(x[0], x[1]) | (pk_e4m3(x[2], x[3]) << 16);
    unsigned int hi = pk_e4m3(x[4], x[5]) | (pk_e4m3(x[6], x[7]) << 16);
    *(uint2*)(qkv8 + idx) = make_uint2(lo, hi);
}

__global__ __launch_bounds__(256) void attn_kernel(
    const unsigned char* __restrict__ q8,
    const unsigned char* __restrict__ k8,
    const unsigned char* __restrict__ v8,
    const float* __restrict__ scales,
    unsigned short* __restrict__ attn_out)
{
    __shared__ __align__(16) unsigned char lK [64 * LKSTR];
    __shared__ __align__(16) unsigned char lVt[128 * 68];
    __shared__ __align__(16) unsigned char lP [8 * 16 * 68];

    const int bh   = blockIdx.y;
    const int q0   = blockIdx.x * 128;
    const int tid  = threadIdx.x;
    const int wave = tid >> 5;
    const int lane = tid & 31;
    const int hl   = lane >> 4;
    const int ln   = lane & 15;

    const float sQK = scales[0] * scales[1] * 0.08838834764831845f;
    const float sV  = scales[2];

    const unsigned char* qBase = q8 + (size_t)bh * SEQ * HD;
    const unsigned char* kBase = k8 + (size_t)bh * SEQ * HD;
    const unsigned char* vBase = v8 + (size_t)bh * SEQ * HD;

    Frag16 qf;
    {
        const unsigned char* qrow = qBase + (size_t)(q0 + wave * 16 + ln) * HD;
#pragma unroll
        for (int v = 0; v < 16; ++v) {
            int off = ((v & 1) * 4) + (((v >> 1) & 1) * 16) + (((v >> 2) & 1) * 32)
                    + (hl ? 8 : 0) + ((v >> 3) * 64);
            qf.u[v] = *(const unsigned int*)(qrow + off);
        }
    }

    float mrow[8], lrow[8];
    AccF O[8];
#pragma unroll
    for (int v = 0; v < 8; ++v) { mrow[v] = -1e30f; lrow[v] = 0.f; }
#pragma unroll
    for (int nt = 0; nt < 8; ++nt)
#pragma unroll
        for (int j = 0; j < 8; ++j) O[nt].f[j] = 0.f;

    for (int kb = 0; kb < SEQ; kb += 64) {
#if HAVE_ASYNC_LDS
#pragma unroll
        for (int j = 0; j < 2; ++j) {
            int c    = tid + 256 * j;
            int kr   = c >> 3, cp = c & 7;
            async_cp16(lK + kr * LKSTR + cp * 16,
                       kBase + (size_t)(kb + kr) * HD + cp * 16);
        }
#else
#pragma unroll
        for (int j = 0; j < 8; ++j) {
            int p  = tid + 256 * j;
            int kr = p >> 5, dc = p & 31;
            unsigned int val = *(const unsigned int*)(kBase + (size_t)(kb + kr) * HD + dc * 4);
            *(unsigned int*)(lK + kr * LKSTR + dc * 4) = val;
        }
#endif
#pragma unroll
        for (int j = 0; j < 8; ++j) {
            int p  = tid + 256 * j;
            int kr = p >> 5, dc = p & 31;
            unsigned int val = *(const unsigned int*)(vBase + (size_t)(kb + kr) * HD + dc * 4);
            lVt[(dc * 4 + 0) * 68 + kr] = (unsigned char)(val);
            lVt[(dc * 4 + 1) * 68 + kr] = (unsigned char)(val >> 8);
            lVt[(dc * 4 + 2) * 68 + kr] = (unsigned char)(val >> 16);
            lVt[(dc * 4 + 3) * 68 + kr] = (unsigned char)(val >> 24);
        }
        if (kb + 64 < SEQ) {
            __builtin_prefetch(kBase + (size_t)(kb + 64 + (tid >> 2)) * HD + (tid & 3) * 32, 0, 0);
            __builtin_prefetch(vBase + (size_t)(kb + 64 + (tid >> 2)) * HD + (tid & 3) * 32, 0, 0);
        }
        wait_async();
        __syncthreads();

        AccF Sc[4];
#pragma unroll
        for (int t = 0; t < 4; ++t) {
            Frag16 bfg;
            int krow = t * 16 + ln;
#pragma unroll
            for (int v = 0; v < 16; ++v) {
                int d = ((v & 1) * 4) + (((v >> 1) & 1) * 16) + (((v >> 2) & 1) * 32) + (hl ? 8 : 0) + ((v >> 3) * 64);
                bfg.u[v] = *(const unsigned int*)(lK + krow * LKSTR + d);
            }
            v8f z = {0.f, 0.f, 0.f, 0.f, 0.f, 0.f, 0.f, 0.f};
            Sc[t].v = __builtin_amdgcn_wmma_f32_16x16x128_fp8_fp8(
                qf.v, bfg.v, (short)0, z, false, false);
        }

        float corr[8];
#pragma unroll
        for (int v = 0; v < 8; ++v) {
            float mx = -1e30f;
#pragma unroll
            for (int t = 0; t < 4; ++t) { Sc[t].f[v] *= sQK; mx = fmaxf(mx, Sc[t].f[v]); }
#pragma unroll
            for (int msk = 8; msk >= 1; msk >>= 1)
                mx = fmaxf(mx, __shfl_xor(mx, msk, 16));
            float mnew = fmaxf(mrow[v], mx);
            corr[v] = __expf(mrow[v] - mnew);
            mrow[v] = mnew;
            float ps = 0.f;
#pragma unroll
            for (int t = 0; t < 4; ++t) {
                float p = __expf(Sc[t].f[v] - mnew);
                Sc[t].f[v] = p; ps += p;
            }
#pragma unroll
            for (int msk = 8; msk >= 1; msk >>= 1)
                ps += __shfl_xor(ps, msk, 16);
            lrow[v] = lrow[v] * corr[v] + ps;
            int row = v + hl * 8;
            unsigned int a01 = pk_e4m3(Sc[0].f[v], Sc[1].f[v]);
            unsigned int a23 = pk_e4m3(Sc[2].f[v], Sc[3].f[v]);
            lP[wave * 1088 + row * 68 +  0 + ln] = (unsigned char)(a01);
            lP[wave * 1088 + row * 68 + 16 + ln] = (unsigned char)(a01 >> 8);
            lP[wave * 1088 + row * 68 + 32 + ln] = (unsigned char)(a23);
            lP[wave * 1088 + row * 68 + 48 + ln] = (unsigned char)(a23 >> 8);
        }
#pragma unroll
        for (int nt = 0; nt < 8; ++nt)
#pragma unroll
            for (int v = 0; v < 8; ++v) O[nt].f[v] *= corr[v];

        Frag8 pf;
#pragma unroll
        for (int v = 0; v < 8; ++v) {
            int koff = ((v & 1) * 4) + (((v >> 1) & 1) * 16) + (((v >> 2) & 1) * 32)
                     + (hl ? 8 : 0);
            pf.u[v] = *(const unsigned int*)(lP + wave * 1088 + ln * 68 + koff);
        }
#pragma unroll
        for (int nt = 0; nt < 8; ++nt) {
            Frag8 vf;
            int d = nt * 16 + ln;
#pragma unroll
            for (int v = 0; v < 8; ++v) {
                int k = ((v & 1) * 4) + (((v >> 1) & 1) * 16) + (((v >> 2) & 1) * 32) + (hl ? 8 : 0);
                vf.u[v] = *(const unsigned int*)(lVt + d * 68 + k);
            }
            O[nt].v = __builtin_amdgcn_wmma_f32_16x16x64_fp8_fp8(
                pf.v, vf.v, (short)0, O[nt].v, false, false);
        }
        __syncthreads();
    }

    const int bb = bh >> 4, h = bh & 15;
#pragma unroll
    for (int v = 0; v < 8; ++v) {
        float inv = sV / lrow[v];
        int s = q0 + wave * 16 + v + hl * 8;
        unsigned short* orow = attn_out + ((size_t)bb * SEQ + s) * HID + h * HD;
#pragma unroll
        for (int nt = 0; nt < 8; ++nt)
            orow[nt * 16 + ln] = f32_to_bf16(O[nt].f[v] * inv);
    }
}

__global__ __launch_bounds__(256) void gemm_out_kernel(
    const unsigned short* __restrict__ A,
    const float* __restrict__ W,
    const float* __restrict__ bias,
    float* __restrict__ Out)
{
    __shared__ __align__(16) unsigned char lA8[128 * LASTR];
    __shared__ unsigned int lB[128 * 17];

    const int m0 = blockIdx.x * 128;
    const int n0 = blockIdx.y * 128;
    const int tid  = threadIdx.x;
    const int wave = tid >> 5;
    const int lane = tid & 31;
    const int hl   = lane >> 4;
    const int ln   = lane & 15;

    AccF acc[8];
#pragma unroll
    for (int nt = 0; nt < 8; ++nt)
#pragma unroll
        for (int j = 0; j < 8; ++j) acc[nt].f[j] = 0.f;

    const unsigned int* A32 = (const unsigned int*)A;
    for (int k0 = 0; k0 < HID; k0 += 32) {
#if HAVE_ASYNC_LDS
#pragma unroll
        for (int j = 0; j < 2; ++j) {
            int c   = tid + 256 * j;
            int row = c >> 2, cp = c & 3;
            async_cp16(lA8 + row * LASTR + cp * 16,
                       (const char*)A + (size_t)(m0 + row) * (HID * 2) + k0 * 2 + cp * 16);
        }
#else
#pragma unroll
        for (int j = 0; j < 8; ++j) {
            int p   = tid + 256 * j;
            int row = p >> 4;
            int kp  = p & 15;
            ((unsigned int*)lA8)[row * (LASTR / 4) + kp] =
                A32[((size_t)(m0 + row) * HID + k0) / 2 + kp];
        }
#endif
#pragma unroll
        for (int j = 0; j < 8; ++j) {
            int p  = tid + 256 * j;
            int n  = p & 127;
            int kp = p >> 7;
            float b0 = W[(size_t)(k0 + 2 * kp)     * HID + n0 + n];
            float b1 = W[(size_t)(k0 + 2 * kp + 1) * HID + n0 + n];
            lB[n * 17 + kp] = (unsigned int)f32_to_bf16(b0) |
                              ((unsigned int)f32_to_bf16(b1) << 16);
        }
        if (k0 + 32 < HID)
            __builtin_prefetch(W + (size_t)(k0 + 32 + (tid >> 4)) * HID + n0 + (tid & 15) * 8, 0, 0);
        wait_async();
        __syncthreads();

        FragB16 af;
#pragma unroll
        for (int v = 0; v < 8; ++v) {
            int kp = (v & 3) + ((v >> 2) * 8) + hl * 4;
            af.u[v] = *(const unsigned int*)(lA8 + (wave * 16 + ln) * LASTR + kp * 4);
        }
#pragma unroll
        for (int nt = 0; nt < 8; ++nt) {
            FragB16 bfg;
#pragma unroll
            for (int v = 0; v < 8; ++v)
                bfg.u[v] = lB[(nt * 16 + ln) * 17 + (v & 3) + ((v >> 2) * 8) + hl * 4];
            acc[nt].v = __builtin_amdgcn_wmma_f32_16x16x32_bf16(
                false, af.v, false, bfg.v, (short)0, acc[nt].v, false, false);
        }
        __syncthreads();
    }

#pragma unroll
    for (int nt = 0; nt < 8; ++nt) {
        float bo = bias[n0 + nt * 16 + ln];
#pragma unroll
        for (int v = 0; v < 8; ++v) {
            int m = m0 + wave * 16 + v + hl * 8;
            int n = n0 + nt * 16 + ln;
            Out[(size_t)m * HID + n] = acc[nt].f[v] + bo;
        }
    }
}

extern "C" void kernel_launch(void* const* d_in, const int* in_sizes, int n_in,
                              void* d_out, int out_size, void* d_ws, size_t ws_size,
                              hipStream_t stream) {
    const float* x     = (const float*)d_in[0];
    const float* W_qkv = (const float*)d_in[1];
    const float* W_out = (const float*)d_in[2];
    const float* b_out = (const float*)d_in[3];
    float* out = (float*)d_out;

    const size_t NELEM = (size_t)BATCH * NH * SEQ * HD;
    const size_t SZ_BF = NELEM * 2;
    char* w = (char*)d_ws;
    unsigned short* qb = (unsigned short*)(w);
    unsigned short* kb = (unsigned short*)(w + 1 * SZ_BF);
    unsigned short* vb = (unsigned short*)(w + 2 * SZ_BF);
    unsigned char*  q8 = (unsigned char*)(w + 3 * SZ_BF);
    unsigned char*  k8 = q8 + NELEM;
    unsigned char*  v8 = k8 + NELEM;
    unsigned short* ao = (unsigned short*)(w + 3 * SZ_BF + 3 * NELEM);
    unsigned int* amax = (unsigned int*)(w + 4 * SZ_BF + 3 * NELEM);
    float* scales      = (float*)(amax + 4);

    init_kernel<<<1, 32, 0, stream>>>(amax, scales);

    gemm_qkv_kernel<<<dim3(MTOT / 128, NQKV / 128), 256, 0, stream>>>(
        x, W_qkv, qb, kb, vb, amax);

    {
        int blocks = (int)((3 * NELEM) / (256 * 8));
        quant_kernel<<<blocks, 256, 0, stream>>>(qb, q8, amax, scales);
    }

    attn_kernel<<<dim3(SEQ / 128, BATCH * NH), 256, 0, stream>>>(
        q8, k8, v8, scales, ao);

    gemm_out_kernel<<<dim3(MTOT / 128, HID / 128), 256, 0, stream>>>(
        ao, W_out, b_out, out);
}
